// WppNetwork_4097398800391
// MI455X (gfx1250) — compile-verified, matrix operand order checked by reading, not run on hardware
//
#include <hip/hip_runtime.h>
#include <hip/hip_bf16.h>
#include <math.h>

// ---------------- WMMA types / helpers (CDNA5 gfx1250, wave32) ----------------
typedef __attribute__((ext_vector_type(16))) __bf16 bf16x16;
typedef __attribute__((ext_vector_type(8)))  float  v8f;

#if defined(__has_builtin)
#if __has_builtin(__builtin_amdgcn_global_load_async_to_lds_b32) && \
    __has_builtin(__builtin_amdgcn_s_wait_asynccnt)
#define HAVE_ASYNC_LDS 1
#endif
#endif
#ifndef HAVE_ASYNC_LDS
#define HAVE_ASYNC_LDS 0
#endif

#if HAVE_ASYNC_LDS
typedef __attribute__((address_space(1))) int as1_int;
typedef __attribute__((address_space(3))) int as3_int;
#endif

static __device__ inline unsigned short bf_bits(float f) {
  unsigned u = __builtin_bit_cast(unsigned, f);
  unsigned r = (u + 0x7FFFu + ((u >> 16) & 1u)) >> 16;   // round-to-nearest-even
  return (unsigned short)r;
}
static __device__ inline __bf16 f2bf(float f) {
  unsigned short s = bf_bits(f);
  return __builtin_bit_cast(__bf16, s);
}
static __device__ inline __bf16 us2bf(unsigned short s) {
  return __builtin_bit_cast(__bf16, s);
}
static __device__ inline float us2f(unsigned short s) {
  unsigned u = ((unsigned)s) << 16;
  return __builtin_bit_cast(float, u);
}
// compile-time part of the 16-bit A/B fragment K index (ISA 7.12.2):
// k = 8*(lane>>4) + kce(e), kce(e) = e for e<8, e+8 for e>=8
static __device__ inline constexpr int kce(int e) { return e + ((e >= 8) ? 8 : 0); }

static __device__ inline v8f wmma_bf16(bf16x16 a, bf16x16 b, v8f c) {
  return __builtin_amdgcn_wmma_f32_16x16x32_bf16(false, a, false, b, (short)0, c, false, false);
}

// ---------------- small utility kernels ----------------
__global__ void zero_f(float* __restrict__ p, int n) {
  int i = blockIdx.x * 256 + threadIdx.x;
  if (i < n) p[i] = 0.f;
}

__global__ void pack_bf16(const float* __restrict__ s, unsigned short* __restrict__ d, int n) {
  int i = blockIdx.x * 256 + threadIdx.x;
  if (i < n) d[i] = bf_bits(s[i]);
}

__global__ void concat3(const float* __restrict__ a, const float* __restrict__ b,
                        const float* __restrict__ c, float* __restrict__ o) {
  int idx = blockIdx.x * 256 + threadIdx.x;
  if (idx >= 32 * 3 * 40000) return;
  int pos = idx % 40000;
  int ch  = (idx / 40000) % 3;
  int bb  = idx / 120000;
  const float* s = (ch == 0) ? a : (ch == 1) ? b : c;
  o[idx] = s[bb * 40000 + pos];
}

// maxpool on bf16 activations; result of max over bf16 values is exactly bf16
__global__ void maxpool2(const unsigned short* __restrict__ in, unsigned short* __restrict__ out,
                         int C, int H, int W) {
  int idx = blockIdx.x * 256 + threadIdx.x;
  int Ho = H >> 1, Wo = W >> 1;
  int total = 32 * C * Ho * Wo;
  if (idx >= total) return;
  int xo = idx % Wo;
  int yo = (idx / Wo) % Ho;
  int c  = (idx / (Wo * Ho)) % C;
  int b  = idx / (Wo * Ho * C);
  const unsigned short* p = in + (((long long)(b * C + c) * H + yo * 2) * W + xo * 2);
  float m = fmaxf(fmaxf(us2f(p[0]), us2f(p[1])), fmaxf(us2f(p[W]), us2f(p[W + 1])));
  out[idx] = bf_bits(m);   // exact (m is already a bf16 value)
}

// ---------------- conv3x3 SAME + bias + ReLU as implicit-GEMM WMMA ----------------
// Activations are bf16 (NCHW). K ordering tap-major: each K=32 chunk has fixed (r,s).
// Weights pre-packed bf16 in OIHW order; index = (co*CI+cin)*9 + tap.
// block = 128 (4 waves -> 4 cout tiles); B tile staged in LDS (async global->LDS b32,
// 2 bf16 per copy; pixels x0-2..x0+17 so global addresses stay 4B-aligned).
template <int CI, int CO, int HH, int WW>
__global__ __launch_bounds__(128) void conv3x3_wmma(
    const void* __restrict__ in_, const unsigned short* __restrict__ wbf,
    const float* __restrict__ bias, unsigned short* __restrict__ out) {
  constexpr int Ktot = CI * 9;
  constexpr int tilesX = (WW + 15) >> 4;
  const int lane = threadIdx.x & 31;
  const int wave = threadIdx.x >> 5;
  const int y0 = blockIdx.x / tilesX;
  const int x0 = (blockIdx.x % tilesX) << 4;
  const int co0 = (blockIdx.y * 4 + wave) << 4;
  const int b = blockIdx.z;
  const int lh = lane >> 4, ln15 = lane & 15;
  v8f acc = {};

  if constexpr (CI == 3) {
    // first layer: f32 input, Ktot=27, one guarded chunk, no barriers
    const float* inb = (const float*)in_ + (long long)b * 3 * HH * WW;
    if (co0 < CO) {
      bf16x16 a, bm;
#pragma unroll
      for (int e = 0; e < 16; ++e) {
        int k = lh * 8 + kce(e);
        unsigned short aw = 0;
        float bv = 0.f;
        if (k < 27) {
          int cin = k / 9, rs = k - cin * 9, r = rs / 3, s = rs - r * 3;
          aw = wbf[(co0 + ln15) * 27 + k];
          int yy = y0 + r - 1, xx = x0 + ln15 + s - 1;
          if (yy >= 0 && yy < HH && xx >= 0 && xx < WW)
            bv = inb[(cin * HH + yy) * WW + xx];
        }
        a[e]  = us2bf(aw);
        bm[e] = f2bf(bv);
      }
      acc = wmma_bf16(a, bm, acc);
    }
  } else {
    const unsigned short* inb = (const unsigned short*)in_ + (long long)b * CI * HH * WW;
    __shared__ unsigned short lds_in[32][20];   // [channel][pixel x0-2 .. x0+17]
#pragma unroll
    for (int tap = 0; tap < 9; ++tap) {
      const int r = tap / 3, s = tap - r * 3;
      const int yy = y0 + r - 1;
      const bool interior = (yy >= 0) && (yy < HH) && (x0 >= 2) && (x0 + 17 < WW);
      for (int cin0 = 0; cin0 < CI; cin0 += 32) {
        __syncthreads();   // previous chunk fully consumed
        if (interior) {
          // 320 b32 copies (2 bf16 each): idx -> c = idx&31, pair j = idx>>5
#if HAVE_ASYNC_LDS
#pragma unroll
          for (int it = 0; it < 3; ++it) {
            int idx = threadIdx.x + it * 128;
            if (idx < 320) {
              int c = idx & 31, j = idx >> 5;
              const unsigned short* gp =
                  inb + ((long long)(cin0 + c) * HH + yy) * WW + (x0 - 2 + 2 * j);
              __builtin_amdgcn_global_load_async_to_lds_b32(
                  (as1_int*)gp, (as3_int*)&lds_in[c][2 * j], 0, 0);
            }
          }
          __builtin_amdgcn_s_wait_asynccnt(0);
#else
#pragma unroll
          for (int it = 0; it < 3; ++it) {
            int idx = threadIdx.x + it * 128;
            if (idx < 320) {
              int c = idx & 31, j = idx >> 5;
              const unsigned short* gp =
                  inb + ((long long)(cin0 + c) * HH + yy) * WW + (x0 - 2 + 2 * j);
              *(int*)&lds_in[c][2 * j] = *(const int*)gp;
            }
          }
#endif
        } else {
#pragma unroll
          for (int it = 0; it < 5; ++it) {
            int idx = threadIdx.x + it * 128;
            int c = idx & 31, i = idx >> 5;    // i in 0..19 for idx<640
            if (i < 20) {
              int xx = x0 - 2 + i;
              unsigned short v = 0;
              if (yy >= 0 && yy < HH && xx >= 0 && xx < WW)
                v = inb[((long long)(cin0 + c) * HH + yy) * WW + xx];
              lds_in[c][i] = v;
            }
          }
        }
        __syncthreads();
        {
          const unsigned short* wrow =
              wbf + (long long)(co0 + ln15) * Ktot + (long long)cin0 * 9 + tap + lh * 72;
          const unsigned short* lrow = &lds_in[lh * 8][ln15 + s + 1];
          bf16x16 a, bm;
#pragma unroll
          for (int e = 0; e < 16; ++e) {
            a[e]  = us2bf(wrow[kce(e) * 9]);
            bm[e] = us2bf(lrow[kce(e) * 20]);
          }
          acc = wmma_bf16(a, bm, acc);
        }
      }
    }
  }

  if (co0 < CO) {
#pragma unroll
    for (int j = 0; j < 8; ++j) {
      int co = co0 + lh * 8 + j;
      int xx = x0 + ln15;
      if (xx < WW) {
        float v = acc[j] + bias[co];
        out[((long long)(b * CO + co) * HH + y0) * WW + xx] = bf_bits(v > 0.f ? v : 0.f);
      }
    }
  }
}

// ---------------- one timestep of the shared 2-layer tanh RNN (B=32, HID=256) ----------------
// 512 threads = 16 waves; wave w -> N-tile w; 2 M-tiles (batch rows).
// Weights AND hidden-state inputs are pre-packed bf16; outputs written f32 + bf16.
__global__ __launch_bounds__(512) void rnn_step(
    const float* __restrict__ xin, int xstride,
    const float* __restrict__ W0i, const float* __restrict__ b0i,
    const unsigned short* __restrict__ W0h, const float* __restrict__ b0h,
    const unsigned short* __restrict__ W1i, const float* __restrict__ b1i,
    const unsigned short* __restrict__ W1h, const float* __restrict__ b1h,
    const unsigned short* __restrict__ h0b_in, const unsigned short* __restrict__ h1b_in,
    float* __restrict__ h0_out, float* __restrict__ h1_out,
    unsigned short* __restrict__ h0b_out, unsigned short* __restrict__ h1b_out) {
  __shared__ unsigned short h0s[32 * 256];   // h0^new staged as bf16 for layer-1 A
  const int lane = threadIdx.x & 31, wave = threadIdx.x >> 5;
  const int lh = lane >> 4, ln15 = lane & 15;
  const int n0 = wave * 16;

  const unsigned short* w0h_r = W0h + (n0 + ln15) * 256 + lh * 8;
  const unsigned short* w1i_r = W1i + (n0 + ln15) * 256 + lh * 8;
  const unsigned short* w1h_r = W1h + (n0 + ln15) * 256 + lh * 8;
  const unsigned short* h0r0 = h0b_in + ln15 * 256 + lh * 8;
  const unsigned short* h0r1 = h0b_in + (16 + ln15) * 256 + lh * 8;
  const unsigned short* h1r0 = h1b_in + ln15 * 256 + lh * 8;
  const unsigned short* h1r1 = h1b_in + (16 + ln15) * 256 + lh * 8;

  // layer 0: h0n = tanh(x@W0i.T + b0i + h0@W0h.T + b0h)
  v8f acc0 = {}, acc1 = {};
  for (int kc = 0; kc < 256; kc += 32) {
    bf16x16 a0, a1, bw;
#pragma unroll
    for (int e = 0; e < 16; ++e) {
      int o = kc + kce(e);
      a0[e] = us2bf(h0r0[o]);
      a1[e] = us2bf(h0r1[o]);
      bw[e] = us2bf(w0h_r[o]);
    }
    acc0 = wmma_bf16(a0, bw, acc0);
    acc1 = wmma_bf16(a1, bw, acc1);
  }
#pragma unroll
  for (int j = 0; j < 8; ++j) {
    int n = n0 + ln15;
    int m0 = lh * 8 + j, m1 = 16 + lh * 8 + j;
    float s0v = acc0[j] + b0i[n] + b0h[n];
    float s1v = acc1[j] + b0i[n] + b0h[n];
#pragma unroll
    for (int k = 0; k < 2; ++k) {
      float w = W0i[n * 2 + k];
      s0v += xin[m0 * xstride + k] * w;
      s1v += xin[m1 * xstride + k] * w;
    }
    s0v = tanhf(s0v); s1v = tanhf(s1v);
    h0_out[m0 * 256 + n] = s0v;  h0_out[m1 * 256 + n] = s1v;
    unsigned short u0 = bf_bits(s0v), u1 = bf_bits(s1v);
    h0b_out[m0 * 256 + n] = u0;  h0b_out[m1 * 256 + n] = u1;
    h0s[m0 * 256 + n] = u0;
    h0s[m1 * 256 + n] = u1;
  }
  __syncthreads();

  // layer 1: h1n = tanh(h0n@W1i.T + b1i + h1@W1h.T + b1h)
  const unsigned short* h0s_r0 = &h0s[ln15 * 256 + lh * 8];
  const unsigned short* h0s_r1 = &h0s[(16 + ln15) * 256 + lh * 8];
  v8f c0 = {}, c1 = {};
  for (int kc = 0; kc < 256; kc += 32) {
    bf16x16 ah0, ah1, bwi, a0, a1, bwh;
#pragma unroll
    for (int e = 0; e < 16; ++e) {
      int o = kc + kce(e);
      ah0[e] = us2bf(h0s_r0[o]);
      ah1[e] = us2bf(h0s_r1[o]);
      bwi[e] = us2bf(w1i_r[o]);
      a0[e]  = us2bf(h1r0[o]);
      a1[e]  = us2bf(h1r1[o]);
      bwh[e] = us2bf(w1h_r[o]);
    }
    c0 = wmma_bf16(ah0, bwi, c0);  c1 = wmma_bf16(ah1, bwi, c1);
    c0 = wmma_bf16(a0,  bwh, c0);  c1 = wmma_bf16(a1,  bwh, c1);
  }
#pragma unroll
  for (int j = 0; j < 8; ++j) {
    int n = n0 + ln15;
    int m0 = lh * 8 + j, m1 = 16 + lh * 8 + j;
    float t0 = tanhf(c0[j] + b1i[n] + b1h[n]);
    float t1 = tanhf(c1[j] + b1i[n] + b1h[n]);
    h1_out[m0 * 256 + n] = t0;  h1_out[m1 * 256 + n] = t1;
    h1b_out[m0 * 256 + n] = bf_bits(t0);
    h1b_out[m1 * 256 + n] = bf_bits(t1);
  }
}

// ---------------- decode (HID->2) and broadcast to all 6 (identical) modes ----------------
__global__ __launch_bounds__(64) void decode_bcast(
    const float* __restrict__ h1, const float* __restrict__ decW,
    const float* __restrict__ decb, float* __restrict__ s_out,
    float* __restrict__ out, int t) {
  int tid = threadIdx.x;
  int b = tid >> 1, c = tid & 1;
  float acc = decb[c];
  for (int k = 0; k < 256; ++k) acc += h1[b * 256 + k] * decW[c * 256 + k];
  s_out[b * 2 + c] = acc;
  for (int mode = 0; mode < 6; ++mode)
    out[((b * 6 + mode) * 12 + t) * 2 + c] = acc;
}

// ---------------- build y = concat([h0,h1], flatten(xenc_bf16)) : (32,627,256) f32 ----------------
__global__ void build_y(const float* __restrict__ h0, const float* __restrict__ h1,
                        const unsigned short* __restrict__ xenc, float* __restrict__ y) {
  int idx = blockIdx.x * 256 + threadIdx.x;
  if (idx >= 32 * 627 * 256) return;
  int d = idx & 255;
  int r = (idx >> 8) % 627;
  int b = idx / (627 * 256);
  float v;
  if (r == 0)      v = h0[b * 256 + d];
  else if (r == 1) v = h1[b * 256 + d];
  else             v = us2f(xenc[(long long)(b * 256 + d) * 625 + (r - 2)]);
  y[idx] = v;
}

// ---------------- row layernorm -> two bf16 outputs (for K and V projections) ----------------
__global__ __launch_bounds__(256) void ln_rows2(
    const float* __restrict__ y,
    const float* __restrict__ gk, const float* __restrict__ bk, unsigned short* __restrict__ klnb,
    const float* __restrict__ gv, const float* __restrict__ bv, unsigned short* __restrict__ vlnb) {
  __shared__ float red[256];
  __shared__ float mu, rs;
  int row = blockIdx.x, d = threadIdx.x;
  float x = y[(long long)row * 256 + d];
  red[d] = x; __syncthreads();
  for (int s = 128; s > 0; s >>= 1) { if (d < s) red[d] += red[d + s]; __syncthreads(); }
  if (d == 0) mu = red[0] * (1.f / 256.f);
  __syncthreads();
  float xm = x - mu;
  red[d] = xm * xm; __syncthreads();
  for (int s = 128; s > 0; s >>= 1) { if (d < s) red[d] += red[d + s]; __syncthreads(); }
  if (d == 0) rs = rsqrtf(red[0] * (1.f / 256.f) + 1e-5f);
  __syncthreads();
  float xn = xm * rs;
  klnb[(long long)row * 256 + d] = bf_bits(xn * gk[d] + bk[d]);
  vlnb[(long long)row * 256 + d] = bf_bits(xn * gv[d] + bv[d]);
}

// ---------------- C(M,256) = A_bf16(M,256) @ W_bf16.T + bias ; one wave per 16x16 tile ----------------
__global__ __launch_bounds__(32) void gemm_rows_wmma(
    const unsigned short* __restrict__ A, const unsigned short* __restrict__ Wt,
    const float* __restrict__ bias, float* __restrict__ C, int Mrows) {
  int lane = threadIdx.x;
  int lh = lane >> 4, ln15 = lane & 15;
  int mt = blockIdx.x, n0 = blockIdx.y * 16;
  int row = mt * 16 + ln15;
  if (row >= Mrows) row = Mrows - 1;
  const unsigned short* ar = A + (long long)row * 256 + lh * 8;
  const unsigned short* wr = Wt + (n0 + ln15) * 256 + lh * 8;
  v8f acc = {};
  for (int kc = 0; kc < 256; kc += 32) {
    bf16x16 a, bw;
#pragma unroll
    for (int e = 0; e < 16; ++e) {
      int o = kc + kce(e);
      a[e]  = us2bf(ar[o]);
      bw[e] = us2bf(wr[o]);
    }
    acc = wmma_bf16(a, bw, acc);
  }
#pragma unroll
  for (int j = 0; j < 8; ++j) {
    int orow = mt * 16 + lh * 8 + j, n = n0 + ln15;
    if (orow < Mrows) C[(long long)orow * 256 + n] = acc[j] + bias[n];
  }
}

// ---------------- Q projection: only rows 0..1 of y are needed ----------------
__global__ __launch_bounds__(256) void q_proj(
    const float* __restrict__ y, const float* __restrict__ g, const float* __restrict__ bb,
    const float* __restrict__ Wq, const float* __restrict__ bq, float* __restrict__ q) {
  __shared__ float t[256];
  __shared__ float red[256];
  __shared__ float mu, rs;
  int b = blockIdx.x, r = blockIdx.y, d = threadIdx.x;
  const float* yr = y + (long long)(b * 627 + r) * 256;
  float x = yr[d];
  red[d] = x; __syncthreads();
  for (int s = 128; s > 0; s >>= 1) { if (d < s) red[d] += red[d + s]; __syncthreads(); }
  if (d == 0) mu = red[0] * (1.f / 256.f);
  __syncthreads();
  float xm = x - mu;
  red[d] = xm * xm; __syncthreads();
  for (int s = 128; s > 0; s >>= 1) { if (d < s) red[d] += red[d + s]; __syncthreads(); }
  if (d == 0) rs = rsqrtf(red[0] * (1.f / 256.f) + 1e-5f);
  __syncthreads();
  t[d] = (xm * rs) * g[d] + bb[d];
  __syncthreads();
  float acc = bq[d];
  for (int k = 0; k < 256; ++k) acc += t[k] * Wq[d * 256 + k];
  q[(b * 2 + r) * 256 + d] = acc;
}

// ---------------- attention: per (b, head), 2 query rows x 627 keys ----------------
__global__ __launch_bounds__(256) void attn_small(
    const float* __restrict__ q, const float* __restrict__ kall,
    const float* __restrict__ vall, float* __restrict__ z) {
  __shared__ float sc[2][627];
  __shared__ float smax[2], ssum[2];
  __shared__ float red0[256], red1[256];
  int b = blockIdx.x, hd = blockIdx.y, t = threadIdx.x;
  const float* qb = q + (b * 2) * 256 + hd * 64;
  for (int key = t; key < 627; key += 256) {
    const float* kp = kall + (long long)(b * 627 + key) * 256 + hd * 64;
    float d0 = 0.f, d1 = 0.f;
    for (int d = 0; d < 64; ++d) { float kv = kp[d]; d0 += qb[d] * kv; d1 += qb[256 + d] * kv; }
    sc[0][key] = d0 * 0.125f;
    sc[1][key] = d1 * 0.125f;
  }
  __syncthreads();
  if (t < 2) {
    float m = -1e30f;
    for (int k = 0; k < 627; ++k) m = fmaxf(m, sc[t][k]);
    smax[t] = m;
  }
  __syncthreads();
  float p0 = 0.f, p1 = 0.f;
  for (int key = t; key < 627; key += 256) {
    float e0 = expf(sc[0][key] - smax[0]); sc[0][key] = e0; p0 += e0;
    float e1 = expf(sc[1][key] - smax[1]); sc[1][key] = e1; p1 += e1;
  }
  red0[t] = p0; red1[t] = p1; __syncthreads();
  for (int s = 128; s > 0; s >>= 1) {
    if (t < s) { red0[t] += red0[t + s]; red1[t] += red1[t + s]; }
    __syncthreads();
  }
  if (t == 0) { ssum[0] = red0[0]; ssum[1] = red1[0]; }
  __syncthreads();
  if (t < 128) {
    int row = t >> 6, d = t & 63;
    float acc = 0.f;
    for (int key = 0; key < 627; ++key)
      acc += sc[row][key] * vall[(long long)(b * 627 + key) * 256 + hd * 64 + d];
    z[(b * 2 + row) * 256 + hd * 64 + d] = acc / ssum[row];
  }
}

// ---------------- residual + LN + MLP(256->512 gelu ->256) + residual ----------------
__global__ __launch_bounds__(256) void mlp_resid(
    const float* __restrict__ z, const float* __restrict__ y,
    const float* __restrict__ g, const float* __restrict__ bb,
    const float* __restrict__ W1, const float* __restrict__ b1,
    const float* __restrict__ W2, const float* __restrict__ b2,
    float* __restrict__ hout0, float* __restrict__ hout1,
    unsigned short* __restrict__ houtb0, unsigned short* __restrict__ houtb1) {
  __shared__ float zres[256], t[256], u[512], red[256];
  __shared__ float mu, rs;
  int b = blockIdx.x, r = blockIdx.y, d = threadIdx.x;
  float zr = z[(b * 2 + r) * 256 + d] + y[(long long)(b * 627 + r) * 256 + d];
  zres[d] = zr;
  red[d] = zr; __syncthreads();
  for (int s = 128; s > 0; s >>= 1) { if (d < s) red[d] += red[d + s]; __syncthreads(); }
  if (d == 0) mu = red[0] * (1.f / 256.f);
  __syncthreads();
  float xm = zr - mu;
  red[d] = xm * xm; __syncthreads();
  for (int s = 128; s > 0; s >>= 1) { if (d < s) red[d] += red[d + s]; __syncthreads(); }
  if (d == 0) rs = rsqrtf(red[0] * (1.f / 256.f) + 1e-5f);
  __syncthreads();
  t[d] = (xm * rs) * g[d] + bb[d];
  __syncthreads();
  for (int jj = 0; jj < 2; ++jj) {
    int j = d + jj * 256;
    float a = b1[j];
    for (int k = 0; k < 256; ++k) a += t[k] * W1[j * 256 + k];
    u[j] = 0.5f * a * (1.f + erff(a * 0.70710678118654752f));   // exact gelu
  }
  __syncthreads();
  float o = b2[d];
  for (int j = 0; j < 512; ++j) o += u[j] * W2[d * 512 + j];
  float hv = o + zres[d];
  if (r == 0) { hout0[b * 256 + d] = hv; houtb0[b * 256 + d] = bf_bits(hv); }
  else        { hout1[b * 256 + d] = hv; houtb1[b * 256 + d] = bf_bits(hv); }
}

// ---------------- host orchestration ----------------
extern "C" void kernel_launch(void* const* d_in, const int* in_sizes, int n_in,
                              void* d_out, int out_size, void* d_ws, size_t ws_size,
                              hipStream_t stream) {
  (void)in_sizes; (void)n_in; (void)out_size; (void)ws_size;
  const float* road = (const float*)d_in[0];
  const float* veh  = (const float*)d_in[1];
  const float* ctr  = (const float*)d_in[2];
  const float* pc   = (const float*)d_in[3];
  // params['attn'] sorted: W1 W2 Wk Wq Wv b1 b2 bk bq bv ln_k_b ln_k_g ln_pre_b ln_pre_g ln_q_b ln_q_g ln_v_b ln_v_g
  const float* aW1 = (const float*)d_in[4];
  const float* aW2 = (const float*)d_in[5];
  const float* aWk = (const float*)d_in[6];
  const float* aWq = (const float*)d_in[7];
  const float* aWv = (const float*)d_in[8];
  const float* ab1 = (const float*)d_in[9];
  const float* ab2 = (const float*)d_in[10];
  const float* abk = (const float*)d_in[11];
  const float* abq = (const float*)d_in[12];
  const float* abv = (const float*)d_in[13];
  const float* ln_k_b  = (const float*)d_in[14];
  const float* ln_k_g  = (const float*)d_in[15];
  const float* ln_pre_b= (const float*)d_in[16];
  const float* ln_pre_g= (const float*)d_in[17];
  const float* ln_q_b  = (const float*)d_in[18];
  const float* ln_q_g  = (const float*)d_in[19];
  const float* ln_v_b  = (const float*)d_in[20];
  const float* ln_v_g  = (const float*)d_in[21];
  const float* cw[6], *cb[6];
  for (int i = 0; i < 6; ++i) { cw[i] = (const float*)d_in[22 + 2 * i]; cb[i] = (const float*)d_in[23 + 2 * i]; }
  const float* decW = (const float*)d_in[34];
  const float* decb = (const float*)d_in[35];
  // params['rnn'] sorted: W0h W0i W1h W1i b0h b0i b1h b1i
  const float* W0h = (const float*)d_in[36];
  const float* W0i = (const float*)d_in[37];
  const float* W1h = (const float*)d_in[38];
  const float* W1i = (const float*)d_in[39];
  const float* b0h = (const float*)d_in[40];
  const float* b0i = (const float*)d_in[41];
  const float* b1h = (const float*)d_in[42];
  const float* b1i = (const float*)d_in[43];

  float* out = (float*)d_out;   // (32, 6, 12, 2)

  // workspace carve-out
  char* p = (char*)d_ws;
  unsigned short* bufA = (unsigned short*)p;  p += (size_t)81920000 * 2;  // bf16 activations ping
  unsigned short* bufB = (unsigned short*)p;  p += (size_t)81920000 * 2;  // bf16 activations pong
  float* xcat = (float*)p;                    p += (size_t)3840000 * 4;   // f32 concat input
  float* ybuf = (float*)p;                    p += (size_t)5136384 * 4;   // (32,627,256)
  unsigned short* klnb = (unsigned short*)p;  p += (size_t)5136384 * 2;
  unsigned short* vlnb = (unsigned short*)p;  p += (size_t)5136384 * 2;
  float* kall = (float*)p;                    p += (size_t)5136384 * 4;
  float* vall = (float*)p;                    p += (size_t)5136384 * 4;
  float* qb   = (float*)p;                    p += (size_t)16384 * 4;
  float* zat  = (float*)p;                    p += (size_t)16384 * 4;
  // f32 hidden-state buffers (3 rotation slots x 2 layers)
  float* hf[3][2];
  for (int sl = 0; sl < 3; ++sl)
    for (int l = 0; l < 2; ++l) { hf[sl][l] = (float*)p; p += (size_t)8192 * 4; }
  // bf16 hidden-state shadows (slot 0 pair is contiguous for one-shot zeroing)
  unsigned short* hb[3][2];
  for (int sl = 0; sl < 3; ++sl)
    for (int l = 0; l < 2; ++l) { hb[sl][l] = (unsigned short*)p; p += (size_t)8192 * 2; }
  float* s0 = (float*)p;                      p += (size_t)64 * 4;
  float* s1 = (float*)p;                      p += (size_t)64 * 4;
  // pre-packed bf16 weights
  const int cwn[6] = {864, 18432, 73728, 147456, 294912, 589824};
  unsigned short* cwbf[6];
  for (int i = 0; i < 6; ++i) { cwbf[i] = (unsigned short*)p; p += (size_t)cwn[i] * 2; }
  unsigned short* w0h_bf = (unsigned short*)p; p += (size_t)65536 * 2;
  unsigned short* w1i_bf = (unsigned short*)p; p += (size_t)65536 * 2;
  unsigned short* w1h_bf = (unsigned short*)p; p += (size_t)65536 * 2;
  unsigned short* wk_bf  = (unsigned short*)p; p += (size_t)65536 * 2;
  unsigned short* wv_bf  = (unsigned short*)p; p += (size_t)65536 * 2;

  // ---- pack reused weights to bf16 once per launch ----
  for (int i = 0; i < 6; ++i)
    pack_bf16<<<(cwn[i] + 255) / 256, 256, 0, stream>>>(cw[i], cwbf[i], cwn[i]);
  pack_bf16<<<256, 256, 0, stream>>>(W0h, w0h_bf, 65536);
  pack_bf16<<<256, 256, 0, stream>>>(W1i, w1i_bf, 65536);
  pack_bf16<<<256, 256, 0, stream>>>(W1h, w1h_bf, 65536);
  pack_bf16<<<256, 256, 0, stream>>>(aWk, wk_bf, 65536);
  pack_bf16<<<256, 256, 0, stream>>>(aWv, wv_bf, 65536);

  // ---- conv encoder (bf16 activations) ----
  concat3<<<(32 * 3 * 40000 + 255) / 256, 256, 0, stream>>>(road, veh, ctr, xcat);
  conv3x3_wmma<3, 32, 200, 200><<<dim3(13 * 200, 1, 32), 128, 0, stream>>>(xcat, cwbf[0], cb[0], bufA);
  conv3x3_wmma<32, 64, 200, 200><<<dim3(13 * 200, 1, 32), 128, 0, stream>>>(bufA, cwbf[1], cb[1], bufB);
  maxpool2<<<(32 * 64 * 100 * 100 + 255) / 256, 256, 0, stream>>>(bufB, bufA, 64, 200, 200);
  conv3x3_wmma<64, 128, 100, 100><<<dim3(7 * 100, 2, 32), 128, 0, stream>>>(bufA, cwbf[2], cb[2], bufB);
  conv3x3_wmma<128, 128, 100, 100><<<dim3(7 * 100, 2, 32), 128, 0, stream>>>(bufB, cwbf[3], cb[3], bufA);
  maxpool2<<<(32 * 128 * 50 * 50 + 255) / 256, 256, 0, stream>>>(bufA, bufB, 128, 100, 100);
  conv3x3_wmma<128, 256, 50, 50><<<dim3(4 * 50, 4, 32), 128, 0, stream>>>(bufB, cwbf[4], cb[4], bufA);
  conv3x3_wmma<256, 256, 50, 50><<<dim3(4 * 50, 4, 32), 128, 0, stream>>>(bufA, cwbf[5], cb[5], bufB);
  maxpool2<<<(32 * 256 * 25 * 25 + 255) / 256, 256, 0, stream>>>(bufB, bufA, 256, 50, 50);
  // bufA now holds xenc (bf16): (32,256,25,25)

  // ---- outer RNN over past trajectory (identical for all 6 modes: compute once) ----
  // slot 0 <-> slot 1 ping-pong; bf16 slot-0 pair zeroed (contiguous: hb[0][0],hb[0][1])
  zero_f<<<(8192 + 255) / 256, 256, 0, stream>>>((float*)hb[0][0], 8192);
  {
    int cur = 0, nxt = 1;
    for (int t = 0; t < 20; ++t) {
      rnn_step<<<1, 512, 0, stream>>>(pc + t * 2, 40,
                                      W0i, b0i, w0h_bf, b0h, w1i_bf, b1i, w1h_bf, b1h,
                                      hb[cur][0], hb[cur][1],
                                      hf[nxt][0], hf[nxt][1], hb[nxt][0], hb[nxt][1]);
      int tt = cur; cur = nxt; nxt = tt;
    }
  }
  // 20 steps -> final state in slot 0
  decode_bcast<<<1, 64, 0, stream>>>(hf[0][1], decW, decb, s0, out, 0);

  // ---- self-attention (2 query rows; K/V over 627 rows via WMMA GEMM) ----
  build_y<<<(32 * 627 * 256 + 255) / 256, 256, 0, stream>>>(hf[0][0], hf[0][1], bufA, ybuf);
  ln_rows2<<<32 * 627, 256, 0, stream>>>(ybuf, ln_k_g, ln_k_b, klnb, ln_v_g, ln_v_b, vlnb);
  gemm_rows_wmma<<<dim3(1254, 16), 32, 0, stream>>>(klnb, wk_bf, abk, kall, 20064);
  gemm_rows_wmma<<<dim3(1254, 16), 32, 0, stream>>>(vlnb, wv_bf, abv, vall, 20064);
  q_proj<<<dim3(32, 2), 256, 0, stream>>>(ybuf, ln_q_g, ln_q_b, aWq, abq, qb);
  attn_small<<<dim3(32, 4), 256, 0, stream>>>(qb, kall, vall, zat);
  // attention result -> slot 2 (f32 + bf16)
  mlp_resid<<<dim3(32, 2), 256, 0, stream>>>(zat, ybuf, ln_pre_g, ln_pre_b,
                                             aW1, ab1, aW2, ab2,
                                             hf[2][0], hf[2][1], hb[2][0], hb[2][1]);

  // ---- inner rollout: 11 steps of (rnn_step on s, decode); rotate slots 2 -> 0 -> 1 -> ... ----
  {
    int cur = 2, nxt = 0, spare = 1;
    float *sc = s0, *sn = s1;
    for (int i = 0; i < 11; ++i) {
      rnn_step<<<1, 512, 0, stream>>>(sc, 2,
                                      W0i, b0i, w0h_bf, b0h, w1i_bf, b1i, w1h_bf, b1h,
                                      hb[cur][0], hb[cur][1],
                                      hf[nxt][0], hf[nxt][1], hb[nxt][0], hb[nxt][1]);
      decode_bcast<<<1, 64, 0, stream>>>(hf[nxt][1], decW, decb, sn, out, 1 + i);
      int t = cur; cur = nxt; nxt = spare; spare = t;
      float* ts = sc; sc = sn; sn = ts;
    }
  }
}
